// ARIG_Fusion_23184233464092
// MI455X (gfx1250) — matrix operand order repaired and checked by reading, not compiled, not run on hardware
//
#include <hip/hip_runtime.h>
#include <hip/hip_bf16.h>

// ---------------------------------------------------------------------------
// ARIG fusion for MI455X (gfx1250, wave32):
//   weights pre-converted to panel-major bf16 once, then streamed to LDS with
//   the Tensor Data Mover (double-buffered, TENSORcnt-synchronized);
//   gates (2 bf16 WMMA GEMMs) -> fuse -> projection (bf16 WMMA GEMM)
//   -> BN stats via LDS + global atomics -> BN apply + 4-step LIF in place.
// ---------------------------------------------------------------------------

typedef __attribute__((ext_vector_type(16))) __bf16        v16bf;
typedef __attribute__((ext_vector_type(8)))  float         v8f;
typedef __attribute__((ext_vector_type(4)))  unsigned int  v4u;
typedef __attribute__((ext_vector_type(8)))  int           v8i;
typedef __attribute__((ext_vector_type(4)))  int           v4i;

#define CDIM  256           // channels (K and N of every GEMM)
#define BM    64            // rows per block
#define MTOT  (128 * 1024)  // T*B*N = 131072 rows
#define KPAN  32            // K panel width (one wmma K step)
#define PANEL (CDIM * KPAN) // 8192 bf16 elements = 16 KB per weight panel
#define WELEM (CDIM * CDIM) // 65536 bf16 elements per converted weight
#define BNC   8388608L      // B*N*C = 32*1024*256

// LDS byte offsets (dynamic LDS base = 0: no static __shared__ in kernel)
#define SW0_OFF (3 * BM * CDIM * 2)        // 98304
#define SW1_OFF (SW0_OFF + PANEL * 2)      // 114688
#define SMEM_BYTES (SW1_OFF + PANEL * 2)   // 131072

#if defined(__gfx1250__) && __has_builtin(__builtin_amdgcn_tensor_load_to_lds) && \
    __has_builtin(__builtin_amdgcn_s_wait_tensorcnt)
#define USE_TDM 1
#else
#define USE_TDM 0
#endif

__device__ __forceinline__ float sigmoidf_(float x) {
    return 1.0f / (1.0f + __expf(-x));
}

// Load a 16x bf16 WMMA fragment from LDS as two 16-byte chunks.
__device__ __forceinline__ v16bf ld_frag16(const __bf16* p, int half_off) {
    union { uint4 u[2]; v16bf v; } r;
    r.u[0] = *reinterpret_cast<const uint4*>(p);
    r.u[1] = *reinterpret_cast<const uint4*>(p + half_off);
    return r.v;
}

// Issue a TDM 1-D tile load: 16 KB contiguous bf16 panel -> LDS[lds_byte_off].
// D# per ISA 08_async_tensor: group0 {count=1, lds_addr, global_addr, type=2},
// group1 {data_size=2B, tensor_dim0 = tile_dim0 = 8192, dim0_stride = 8192}.
// 6-arg builtin form (amdgpu-toolchain clang-23): (v4u, v8i, v4i, v4i, v8i, i32).
__device__ __forceinline__ void issue_tdm_panel(unsigned lds_byte_off, const __bf16* gsrc) {
#if USE_TDM
    const unsigned long long ga = (unsigned long long)gsrc;
    v4u g0 = { 1u,                                  // count=1, is_restore=0, gather=0
               lds_byte_off,                        // lds_addr[31:0]
               (unsigned)ga,                        // global_addr[31:0]
               (unsigned)((ga >> 32) & 0x01FFFFFFu) | (2u << 30) };  // addr[56:32] | type=2
    v8i g1 = { 0x00010000,                          // workgroup_mask=0, data_size=1 (2B)
               0x20000000,                          // tensor_dim0[15:0]=8192 (bits[63:48])
               0x00010000,                          // tensor_dim0 hi=0, tensor_dim1=1
               0x20000000,                          // tensor_dim1 hi=0, tile_dim0=8192
               1,                                   // tile_dim1=1, tile_dim2=0
               PANEL,                               // tensor_dim0_stride lo = 8192
               0, 0 };                              // stride hi / tensor_dim1_stride = 0
    v4i gz4 = { 0, 0, 0, 0 };
    v8i gz8 = { 0, 0, 0, 0, 0, 0, 0, 0 };
    __builtin_amdgcn_tensor_load_to_lds(g0, g1, gz4, gz4, gz8, 0);
#else
    (void)lds_byte_off; (void)gsrc;
#endif
}

// One K panel of the block GEMM: frags from LDS, 2x4 wmma tiles per wave.
__device__ __forceinline__ void compute_panel(const __bf16* sX, const __bf16* sW, int k0,
                                              v8f acc[8], int lane, int mwave, int nwave)
{
    v16bf af[2], bfr[4];
    const int klo  = (lane < 16) ? 0 : 8;    // A 16-bit layout (ISA 7.12.2)
    const int koff = (lane < 16) ? 0 : 8;    // B 16-bit layout
    #pragma unroll
    for (int mt = 0; mt < 2; ++mt) {
        const __bf16* p = sX + (mwave + mt * 16 + (lane & 15)) * CDIM + k0 + klo;
        af[mt] = ld_frag16(p, 16);
    }
    #pragma unroll
    for (int nt = 0; nt < 4; ++nt) {
        const __bf16* p = sW + (nwave + nt * 16 + (lane & 15)) * KPAN + koff;
        bfr[nt] = ld_frag16(p, 16);
    }
    #pragma unroll
    for (int mt = 0; mt < 2; ++mt)
        #pragma unroll
        for (int nt = 0; nt < 4; ++nt)
            acc[mt * 4 + nt] = __builtin_amdgcn_wmma_f32_16x16x32_bf16(
                false, af[mt], false, bfr[nt], (short)0, acc[mt * 4 + nt], false, false);
}

// acc[2x4 tiles] += sX(64x256,bf16) * W^T, W pre-converted bf16 panel-major
// [8][256][32]. TDM path double-buffers panels through sW0/sW1.
__device__ __forceinline__ void gemm_block(const __bf16* sX, __bf16* sW0, __bf16* sW1,
                                           const __bf16* __restrict__ Wb,
                                           v8f acc[8], int lane, int mwave, int nwave)
{
#if USE_TDM
    if (threadIdx.x < 32) issue_tdm_panel(SW0_OFF, Wb);   // prologue: panel 0 -> buf0
    for (int kp = 0; kp < 8; ++kp) {
        const __bf16* sW = (kp & 1) ? sW1 : sW0;
        if (threadIdx.x < 32) {
            if (kp < 7) {
                issue_tdm_panel((kp & 1) ? SW0_OFF : SW1_OFF, Wb + (kp + 1) * PANEL);
                __builtin_amdgcn_s_wait_tensorcnt(1);     // panel kp has landed
            } else {
                __builtin_amdgcn_s_wait_tensorcnt(0);
            }
        }
        __syncthreads();
        compute_panel(sX, sW, kp * KPAN, acc, lane, mwave, nwave);
        __syncthreads();
    }
#else
    for (int kp = 0; kp < 8; ++kp) {
        const __bf16* src = Wb + kp * PANEL + threadIdx.x * KPAN;  // 64 B per thread
        __bf16* dst = sW0 + threadIdx.x * KPAN;
        *reinterpret_cast<uint4*>(dst)      = *reinterpret_cast<const uint4*>(src);
        *reinterpret_cast<uint4*>(dst + 8)  = *reinterpret_cast<const uint4*>(src + 8);
        *reinterpret_cast<uint4*>(dst + 16) = *reinterpret_cast<const uint4*>(src + 16);
        *reinterpret_cast<uint4*>(dst + 24) = *reinterpret_cast<const uint4*>(src + 24);
        __syncthreads();
        compute_panel(sX, sW0, kp * KPAN, acc, lane, mwave, nwave);
        __syncthreads();
    }
    (void)sW1;
#endif
}

// Pre-convert the three fp32 weights to bf16, panel-major [kp][d][kk].
__global__ __launch_bounds__(256) void ARIG_wconv(
    const float* __restrict__ Wa, const float* __restrict__ Wl,
    const float* __restrict__ Wp, __bf16* __restrict__ Wb)
{
    const int flat = blockIdx.x * blockDim.x + threadIdx.x;    // [0, 49152)
    const int w    = flat >> 14;                               // 16384 quads per weight
    const int e    = (flat & 16383) * 4;                       // dst element, kk-aligned
    const int kk   = e & 31;
    const int d    = (e >> 5) & 255;
    const int kp   = e >> 13;
    const float* Wsrc = (w == 0) ? Wa : (w == 1) ? Wl : Wp;
    float4 f = *reinterpret_cast<const float4*>(Wsrc + d * CDIM + kp * KPAN + kk);
    __bf16* dst = Wb + w * WELEM + e;
    dst[0] = (__bf16)f.x; dst[1] = (__bf16)f.y;
    dst[2] = (__bf16)f.z; dst[3] = (__bf16)f.w;
}

__global__ __launch_bounds__(256, 1) void ARIG_fused_gemm(
    const float* __restrict__ xa, const float* __restrict__ xl,
    const float* __restrict__ ba, const float* __restrict__ bl,
    const float* __restrict__ bp, const __bf16* __restrict__ Wb,
    float* __restrict__ gOut, float* __restrict__ gSum, float* __restrict__ gSumSq)
{
    extern __shared__ char smem[];
    __bf16* sA  = reinterpret_cast<__bf16*>(smem);   // 64 x 256 (a, later fused F)
    __bf16* sL  = sA + BM * CDIM;                    // 64 x 256 (l)
    __bf16* sG  = sL + BM * CDIM;                    // 64 x 256 (gate_attn)
    __bf16* sW0 = sG + BM * CDIM;                    // 256 x 32 panel buf0
    __bf16* sW1 = sW0 + PANEL;                       // 256 x 32 panel buf1
    float*  sRed = reinterpret_cast<float*>(sG);     // 512 floats, reused after fuse

    const int tid   = threadIdx.x;
    const int lane  = tid & 31;
    const int wave  = tid >> 5;
    const int mwave = (wave >> 2) * 32;              // 2 row-groups of waves
    const int nwave = (wave & 3) * 64;               // 4 col-groups of waves
    const int mhalf = (lane >= 16) ? 8 : 0;          // C/D layout: M = r (+8 high half)
    const int ncol  = lane & 15;                     // C/D layout: N = lane%16
    const long rowBase = (long)blockIdx.x * BM;

    // ---- stage x_attn / x_lsm block (fp32 -> bf16), coalesced float4 ----
    {
        const float* gxa = xa + rowBase * CDIM;
        const float* gxl = xl + rowBase * CDIM;
        #pragma unroll
        for (int i = 0; i < 16; ++i) {
            const int idx = tid * 4 + i * 1024;
            float4 va = *reinterpret_cast<const float4*>(gxa + idx);
            float4 vl = *reinterpret_cast<const float4*>(gxl + idx);
            sA[idx + 0] = (__bf16)va.x; sA[idx + 1] = (__bf16)va.y;
            sA[idx + 2] = (__bf16)va.z; sA[idx + 3] = (__bf16)va.w;
            sL[idx + 0] = (__bf16)vl.x; sL[idx + 1] = (__bf16)vl.y;
            sL[idx + 2] = (__bf16)vl.z; sL[idx + 3] = (__bf16)vl.w;
        }
    }
    __syncthreads();

    v8f acc[8];
    const v8f vzero = {0.f, 0.f, 0.f, 0.f, 0.f, 0.f, 0.f, 0.f};

    // ===== Phase 1: gate_attn = sigmoid(L*W_lsm^T + b_lsm) -> sG =====
    #pragma unroll
    for (int i = 0; i < 8; ++i) acc[i] = vzero;
    gemm_block(sL, sW0, sW1, Wb + 1 * WELEM, acc, lane, mwave, nwave);
    #pragma unroll
    for (int mt = 0; mt < 2; ++mt)
        #pragma unroll
        for (int nt = 0; nt < 4; ++nt) {
            const int d = nwave + nt * 16 + ncol;
            const float bv = bl[d];
            #pragma unroll
            for (int r = 0; r < 8; ++r) {
                const int m = mwave + mt * 16 + r + mhalf;
                sG[m * CDIM + d] = (__bf16)sigmoidf_(acc[mt * 4 + nt][r] + bv);
            }
        }
    __syncthreads();

    // ===== Phase 2: gate_lsm = sigmoid(A*W_att^T + b_att); F = a*g_attn + l*g_lsm =====
    #pragma unroll
    for (int i = 0; i < 8; ++i) acc[i] = vzero;
    gemm_block(sA, sW0, sW1, Wb + 0 * WELEM, acc, lane, mwave, nwave);
    #pragma unroll
    for (int mt = 0; mt < 2; ++mt)
        #pragma unroll
        for (int nt = 0; nt < 4; ++nt) {
            const int d = nwave + nt * 16 + ncol;
            const float bv = ba[d];
            #pragma unroll
            for (int r = 0; r < 8; ++r) {
                const int m = mwave + mt * 16 + r + mhalf;
                const float g1 = sigmoidf_(acc[mt * 4 + nt][r] + bv); // gates l
                const float g2 = (float)sG[m * CDIM + d];             // gates a
                const float av = (float)sA[m * CDIM + d];
                const float lv = (float)sL[m * CDIM + d];
                sA[m * CDIM + d] = (__bf16)(av * g2 + lv * g1);       // in place, disjoint
            }
        }
    __syncthreads();

    // ===== Phase 3: out = F*W_proj^T + b_proj; emit + BN partial sums =====
    #pragma unroll
    for (int i = 0; i < 8; ++i) acc[i] = vzero;
    gemm_block(sA, sW0, sW1, Wb + 2 * WELEM, acc, lane, mwave, nwave);

    sRed[tid] = 0.f; sRed[tid + 256] = 0.f;   // sG region free now
    __syncthreads();

    #pragma unroll
    for (int mt = 0; mt < 2; ++mt)
        #pragma unroll
        for (int nt = 0; nt < 4; ++nt) {
            const int d = nwave + nt * 16 + ncol;
            const float bv = bp[d];
            float lsum = 0.f, lsq = 0.f;
            #pragma unroll
            for (int r = 0; r < 8; ++r) {
                const int m = mwave + mt * 16 + r + mhalf;
                const float o = acc[mt * 4 + nt][r] + bv;
                gOut[(rowBase + m) * CDIM + d] = o;
                lsum += o; lsq += o * o;
            }
            atomicAdd(&sRed[d], lsum);          // ds_add_f32
            atomicAdd(&sRed[256 + d], lsq);
        }
    __syncthreads();
    atomicAdd(&gSum[tid],   sRed[tid]);         // 2 global atomics per channel per block
    atomicAdd(&gSumSq[tid], sRed[tid + 256]);
}

__global__ void ARIG_bn_stats(const float* __restrict__ gSum, const float* __restrict__ gSumSq,
                              float* __restrict__ mean, float* __restrict__ invstd)
{
    const int d = threadIdx.x;
    const float inv_n = 1.0f / (float)MTOT;
    const float m = gSum[d] * inv_n;
    const float v = gSumSq[d] * inv_n - m * m;
    mean[d] = m;
    invstd[d] = rsqrtf(v + 1e-5f);
}

__global__ __launch_bounds__(256) void ARIG_bn_lif(
    float* __restrict__ io, const float* __restrict__ gamma,
    const float* __restrict__ beta, const float* __restrict__ mean,
    const float* __restrict__ invstd, const float* __restrict__ lifw)
{
    const long flat = ((long)blockIdx.x * blockDim.x + threadIdx.x) * 4;
    if (flat >= BNC) return;
    const int c = (int)(flat & (CDIM - 1));
    const float tinv = sigmoidf_(lifw[0]);
    const float4 ga = *reinterpret_cast<const float4*>(gamma + c);
    const float4 be = *reinterpret_cast<const float4*>(beta + c);
    const float4 mu = *reinterpret_cast<const float4*>(mean + c);
    const float4 is = *reinterpret_cast<const float4*>(invstd + c);
    float v0 = 0.f, v1 = 0.f, v2 = 0.f, v3 = 0.f;
    #pragma unroll
    for (int t = 0; t < 4; ++t) {
        float4 x = *reinterpret_cast<const float4*>(io + (long)t * BNC + flat);
        const float y0 = ga.x * (x.x - mu.x) * is.x + be.x;
        const float y1 = ga.y * (x.y - mu.y) * is.y + be.y;
        const float y2 = ga.z * (x.z - mu.z) * is.z + be.z;
        const float y3 = ga.w * (x.w - mu.w) * is.w + be.w;
        v0 += (y0 - v0) * tinv;  v1 += (y1 - v1) * tinv;
        v2 += (y2 - v2) * tinv;  v3 += (y3 - v3) * tinv;
        const float s0 = (v0 >= 1.0f) ? 1.0f : 0.0f;
        const float s1 = (v1 >= 1.0f) ? 1.0f : 0.0f;
        const float s2 = (v2 >= 1.0f) ? 1.0f : 0.0f;
        const float s3 = (v3 >= 1.0f) ? 1.0f : 0.0f;
        v0 *= (1.0f - s0); v1 *= (1.0f - s1); v2 *= (1.0f - s2); v3 *= (1.0f - s3);
        float4 sv; sv.x = s0; sv.y = s1; sv.z = s2; sv.w = s3;
        *reinterpret_cast<float4*>(io + (long)t * BNC + flat) = sv;  // in place
    }
}

extern "C" void kernel_launch(void* const* d_in, const int* in_sizes, int n_in,
                              void* d_out, int out_size, void* d_ws, size_t ws_size,
                              hipStream_t stream) {
    const float* xa = (const float*)d_in[0];
    const float* xl = (const float*)d_in[1];
    const float* Wa = (const float*)d_in[2];
    const float* ba = (const float*)d_in[3];
    const float* Wl = (const float*)d_in[4];
    const float* bl = (const float*)d_in[5];
    const float* Wp = (const float*)d_in[6];
    const float* bp = (const float*)d_in[7];
    const float* gamma = (const float*)d_in[8];
    const float* beta  = (const float*)d_in[9];
    const float* lifw  = (const float*)d_in[10];

    float* out    = (float*)d_out;        // pre-BN written here, LIF rewrites in place
    float* gSum   = (float*)d_ws;         // [256]
    float* gSumSq = gSum + 256;           // [256]
    float* mean   = gSum + 512;           // [256]
    float* invstd = gSum + 768;           // [256]
    __bf16* Wb    = (__bf16*)((char*)d_ws + 4096);  // 3 x 65536 bf16, panel-major

    (void)hipMemsetAsync(d_ws, 0, 1024 * sizeof(float), stream);
    ARIG_wconv<<<192, 256, 0, stream>>>(Wa, Wl, Wp, Wb);

    (void)hipFuncSetAttribute((const void*)ARIG_fused_gemm,
                              hipFuncAttributeMaxDynamicSharedMemorySize, SMEM_BYTES);
    ARIG_fused_gemm<<<MTOT / BM, 256, SMEM_BYTES, stream>>>(
        xa, xl, ba, bl, bp, Wb, out, gSum, gSumSq);
    ARIG_bn_stats<<<1, 256, 0, stream>>>(gSum, gSumSq, mean, invstd);
    ARIG_bn_lif<<<(int)((BNC / 4) / 256), 256, 0, stream>>>(out, gamma, beta, mean, invstd, lifw);
}
